// MixtralMoE_47244640256292
// MI455X (gfx1250) — compile-verified, matrix operand order checked by reading, not run on hardware
//
#include <hip/hip_runtime.h>
#include <hip/hip_bf16.h>
#include <cstdint>

// Problem constants (B=1)
#define T_TOK 4096
#define HDIM  1024
#define FDIM  3584
#define NEXP  8
#define TOPK  2

// GEMM tiling: 128x128 block tile, K-step 32 (bf16 WMMA K), 256 threads = 8 waves
#define MT 128
#define NT 128
#define KT 32
#define ROWU 40   // ushorts per LDS row: 32 bf16 + 8 pad (80B stride, conflict-free b128 reads)

typedef unsigned short u16;
typedef __bf16 bf16;
typedef bf16  v16bf __attribute__((ext_vector_type(16)));
typedef float v8f   __attribute__((ext_vector_type(8)));
typedef int   v4i   __attribute__((ext_vector_type(4)));

// ---------------------------------------------------------------------------
// Async global->LDS copy (CDNA5 GLOBAL_LOAD_ASYNC_TO_LDS_B128, ASYNCcnt) with
// synchronous fallback if the builtin is not declared by this toolchain.
// Builtin signature (from toolchain diagnostic): (v4i AS1*, v4i AS3*, imm, imm).
// ---------------------------------------------------------------------------
#if defined(__has_builtin)
#if __has_builtin(__builtin_amdgcn_global_load_async_to_lds_b128)
#define HAVE_ASYNC_LDS 1
#endif
#endif

typedef __attribute__((address_space(1))) v4i as1_v4i;
typedef __attribute__((address_space(3))) v4i as3_v4i;

__device__ __forceinline__ void copy16_to_lds(void* lds_dst, const void* gsrc) {
#if defined(HAVE_ASYNC_LDS)
  __builtin_amdgcn_global_load_async_to_lds_b128(
      (as1_v4i*)(const_cast<void*>(gsrc)), (as3_v4i*)lds_dst, 0, 0);
#else
  *(uint4*)lds_dst = *(const uint4*)gsrc;
#endif
}

__device__ __forceinline__ void async_wait0() {
#if defined(HAVE_ASYNC_LDS)
#if __has_builtin(__builtin_amdgcn_s_wait_asynccnt)
  __builtin_amdgcn_s_wait_asynccnt(0);
#else
  asm volatile("s_wait_asynccnt 0x0" ::: "memory");
#endif
#endif
}

union FragU { uint4 u[2]; v16bf v; };

// Load one 16x32 bf16 fragment (A or B^T) from an LDS row.
// Lane layout per CDNA5 ISA: lanes 0-15 hold K 0-7 & 16-23, lanes 16-31 hold K 8-15 & 24-31,
// i.e. 16B chunk at byte 16*half and 16B chunk at byte 32+16*half of the 64B row.
__device__ __forceinline__ v16bf frag_from_lds(const u16* rowbase, int half) {
  FragU f;
  const uint4* p = (const uint4*)(rowbase + 8 * half);
  f.u[0] = p[0];
  f.u[1] = p[2];
  return f.v;
}

// Convert 4 fp32 -> 4 bf16 and store 8B into LDS.
__device__ __forceinline__ void cvt_store4(u16* dst, float4 v) {
  union { bf16 b[4]; uint2 u; } p;
  p.b[0] = (bf16)v.x; p.b[1] = (bf16)v.y; p.b[2] = (bf16)v.z; p.b[3] = (bf16)v.w;
  *(uint2*)dst = p.u;
}

// ---------------------------------------------------------------------------
// 1) Gate: logits = x @ gate_w^T, softmax, top-2, count tokens per expert.
//    One wave (32 lanes) per token; gate_w staged in LDS (32 KB) via async DMA.
// ---------------------------------------------------------------------------
__global__ __launch_bounds__(256) void gate_kernel(
    const float* __restrict__ x, const float* __restrict__ gw,
    int* __restrict__ cnt, int* __restrict__ ti, float* __restrict__ tw)
{
  __shared__ float4 g4[NEXP * 256];   // 8 experts x 1024 floats
  const int tid = threadIdx.x;
#pragma unroll
  for (int i = 0; i < 8; ++i)
    copy16_to_lds(&g4[tid + i * 256], &((const float4*)gw)[tid + i * 256]);
  async_wait0();
  __syncthreads();

  const int wid = tid >> 5, lane = tid & 31;
  const int t = blockIdx.x * 8 + wid;

  float l[NEXP];
#pragma unroll
  for (int e = 0; e < NEXP; ++e) l[e] = 0.f;

  const float4* xr = (const float4*)(x + (size_t)t * HDIM);
#pragma unroll
  for (int c = 0; c < 8; ++c) {
    float4 xv = xr[lane + c * 32];
#pragma unroll
    for (int e = 0; e < NEXP; ++e) {
      float4 gv = g4[e * 256 + lane + c * 32];
      l[e] += xv.x * gv.x + xv.y * gv.y + xv.z * gv.z + xv.w * gv.w;
    }
  }
#pragma unroll
  for (int e = 0; e < NEXP; ++e)
    for (int o = 16; o > 0; o >>= 1) l[e] += __shfl_xor(l[e], o, 32);

  if (lane == 0) {
    float mx = l[0];
#pragma unroll
    for (int e = 1; e < NEXP; ++e) mx = fmaxf(mx, l[e]);
    float p[NEXP], s = 0.f;
#pragma unroll
    for (int e = 0; e < NEXP; ++e) { p[e] = __expf(l[e] - mx); s += p[e]; }
    float inv = 1.f / s;
#pragma unroll
    for (int e = 0; e < NEXP; ++e) p[e] *= inv;
    int i1 = 0;
#pragma unroll
    for (int e = 1; e < NEXP; ++e) if (p[e] > p[i1]) i1 = e;
    int i2 = (i1 == 0) ? 1 : 0;
#pragma unroll
    for (int e = 0; e < NEXP; ++e) if (e != i1 && p[e] > p[i2]) i2 = e;
    ti[t * 2 + 0] = i1; tw[t * 2 + 0] = p[i1];
    ti[t * 2 + 1] = i2; tw[t * 2 + 1] = p[i2];
    atomicAdd(&cnt[i1], 1);
    atomicAdd(&cnt[i2], 1);
  }
}

// 2) Exclusive scan of 8 expert counts.
__global__ void offsets_kernel(const int* __restrict__ cnt, int* __restrict__ off) {
  if (threadIdx.x == 0 && blockIdx.x == 0) {
    int a = 0;
    for (int e = 0; e < NEXP; ++e) { off[e] = a; a += cnt[e]; }
    off[NEXP] = a;
  }
}

// 3) Compact (token, expert) assignments into per-expert contiguous slots.
__global__ __launch_bounds__(256) void build_kernel(
    const int* __restrict__ ti, const float* __restrict__ tw,
    const int* __restrict__ off, int* __restrict__ cur,
    int* __restrict__ slot_of, float* __restrict__ wslot, int* __restrict__ tok_slot)
{
  const int t = blockIdx.x * 256 + threadIdx.x;
  if (t >= T_TOK) return;
#pragma unroll
  for (int k = 0; k < TOPK; ++k) {
    int e = ti[t * 2 + k];
    int s = atomicAdd(&cur[e], 1);
    int slot = off[e] + s;
    slot_of[t * 2 + k] = slot;
    wslot[slot] = tw[t * 2 + k];
    tok_slot[slot] = t;
  }
}

// ---------------------------------------------------------------------------
// 4) FFN up-projections: act = silu(x @ w1^T) * (x @ w3^T), stored bf16.
//    Block = (expert, token-tile 128, f-tile 128). bf16 WMMA, f32 accum.
// ---------------------------------------------------------------------------
__global__ __launch_bounds__(256) void ffn1_kernel(
    const float* __restrict__ x, const float* __restrict__ w1,
    const float* __restrict__ w3, const int* __restrict__ cnt,
    const int* __restrict__ off, const int* __restrict__ tok_slot,
    bf16* __restrict__ act)
{
  const int e  = blockIdx.z;
  const int mt = blockIdx.y;
  const int ft = blockIdx.x;
  const int cnt_e = cnt[e];
  const int m0 = mt * MT;
  if (m0 >= cnt_e) return;                 // uniform early exit
  const int valid = min(MT, cnt_e - m0);
  const int aBase = off[e] + m0;
  const int f0 = ft * NT;
  const float* w1e = w1 + (size_t)e * FDIM * HDIM;
  const float* w3e = w3 + (size_t)e * FDIM * HDIM;

  __shared__ u16 lA[MT * ROWU];
  __shared__ u16 lB1[NT * ROWU];
  __shared__ u16 lB3[NT * ROWU];
  __shared__ int stok[MT];

  const int tid = threadIdx.x;
  if (tid < MT) stok[tid] = (tid < valid) ? tok_slot[aBase + tid] : 0;

  const int wid = tid >> 5, lane = tid & 31;
  const int half = lane >> 4, mrow = lane & 15;
  const int Mb = (wid & 3) * 32, Nb = (wid >> 2) * 64;

  v8f acc1[8], acc3[8];
#pragma unroll
  for (int i = 0; i < 8; ++i)
#pragma unroll
    for (int c = 0; c < 8; ++c) { acc1[i][c] = 0.f; acc3[i][c] = 0.f; }

  for (int kk = 0; kk < HDIM; kk += KT) {
    __syncthreads();
#pragma unroll
    for (int i = 0; i < 4; ++i) {
      int idx = tid + i * 256;             // 1024 float4 slots per tile
      int row = idx >> 3, c4 = idx & 7;
      const float4* sa = (const float4*)(x   + (size_t)stok[row] * HDIM + kk) + c4;
      const float4* s1 = (const float4*)(w1e + (size_t)(f0 + row) * HDIM + kk) + c4;
      const float4* s3 = (const float4*)(w3e + (size_t)(f0 + row) * HDIM + kk) + c4;
      cvt_store4(&lA [row * ROWU + c4 * 4], *sa);
      cvt_store4(&lB1[row * ROWU + c4 * 4], *s1);
      cvt_store4(&lB3[row * ROWU + c4 * 4], *s3);
      if (kk + KT < HDIM) {                // stream next K-slice of weights
        __builtin_prefetch((const void*)(s1 + 8), 0, 1);
        __builtin_prefetch((const void*)(s3 + 8), 0, 1);
      }
    }
    __syncthreads();

    v16bf af[2], bfr[4];
#pragma unroll
    for (int i = 0; i < 2; ++i) af[i] = frag_from_lds(&lA[(Mb + i * 16 + mrow) * ROWU], half);
#pragma unroll
    for (int j = 0; j < 4; ++j) bfr[j] = frag_from_lds(&lB1[(Nb + j * 16 + mrow) * ROWU], half);
#pragma unroll
    for (int i = 0; i < 2; ++i)
#pragma unroll
      for (int j = 0; j < 4; ++j)
        acc1[i * 4 + j] = __builtin_amdgcn_wmma_f32_16x16x32_bf16(
            false, af[i], false, bfr[j], (short)0, acc1[i * 4 + j], false, false);
#pragma unroll
    for (int j = 0; j < 4; ++j) bfr[j] = frag_from_lds(&lB3[(Nb + j * 16 + mrow) * ROWU], half);
#pragma unroll
    for (int i = 0; i < 2; ++i)
#pragma unroll
      for (int j = 0; j < 4; ++j)
        acc3[i * 4 + j] = __builtin_amdgcn_wmma_f32_16x16x32_bf16(
            false, af[i], false, bfr[j], (short)0, acc3[i * 4 + j], false, false);
  }

  // Epilogue: silu(s1)*s3 -> bf16 act rows (fast path for full tiles).
#pragma unroll
  for (int i = 0; i < 2; ++i)
#pragma unroll
    for (int j = 0; j < 4; ++j) {
      v8f s1 = acc1[i * 4 + j], s3 = acc3[i * 4 + j];
      int n = Nb + j * 16 + mrow;
      bf16* arow = act + (size_t)(aBase + Mb + i * 16 + half * 8) * FDIM + (f0 + n);
      if (valid == MT) {
#pragma unroll
        for (int c = 0; c < 8; ++c) {
          float a = s1[c];
          float g = a / (1.0f + __expf(-a)) * s3[c];
          arow[(size_t)c * FDIM] = (bf16)g;
        }
      } else {
#pragma unroll
        for (int c = 0; c < 8; ++c) {
          int m = Mb + i * 16 + half * 8 + c;
          if (m < valid) {
            float a = s1[c];
            float g = a / (1.0f + __expf(-a)) * s3[c];
            arow[(size_t)c * FDIM] = (bf16)g;
          }
        }
      }
    }
}

// ---------------------------------------------------------------------------
// 5) Down-projection: contrib = gate_w * (act @ w2^T). Per-assignment rows,
//    non-atomic stores (deterministic combine below). A-tile staged with
//    async global->LDS DMA (no conversion needed: act is already bf16).
// ---------------------------------------------------------------------------
__global__ __launch_bounds__(256) void ffn2_kernel(
    const bf16* __restrict__ act, const float* __restrict__ w2,
    const int* __restrict__ cnt, const int* __restrict__ off,
    const float* __restrict__ wslot, float* __restrict__ contrib)
{
  const int e  = blockIdx.z;
  const int mt = blockIdx.y;
  const int ht = blockIdx.x;
  const int cnt_e = cnt[e];
  const int m0 = mt * MT;
  if (m0 >= cnt_e) return;
  const int valid = min(MT, cnt_e - m0);
  const int aBase = off[e] + m0;
  const int h0 = ht * NT;
  const float* w2e = w2 + (size_t)e * HDIM * FDIM;

  __shared__ u16 lA[MT * ROWU];
  __shared__ u16 lB[NT * ROWU];
  __shared__ float sw[MT];

  const int tid = threadIdx.x;
  if (tid < MT) sw[tid] = (tid < valid) ? wslot[aBase + tid] : 0.f;

  const int wid = tid >> 5, lane = tid & 31;
  const int half = lane >> 4, mrow = lane & 15;
  const int Mb = (wid & 3) * 32, Nb = (wid >> 2) * 64;

  v8f acc[8];
#pragma unroll
  for (int i = 0; i < 8; ++i)
#pragma unroll
    for (int c = 0; c < 8; ++c) acc[i][c] = 0.f;

  for (int kk = 0; kk < FDIM; kk += KT) {
    __syncthreads();
    // A: bf16 act rows -> LDS via async DMA (512 x 16B, 2/thread)
#pragma unroll
    for (int i = 0; i < 2; ++i) {
      int idx = tid + i * 256;
      int row = idx >> 2, q = idx & 3;
      const uint4* s = (const uint4*)(act + (size_t)(aBase + row) * FDIM + kk) + q;
      copy16_to_lds(&lA[row * ROWU + q * 8], s);
    }
    // B: w2 rows fp32 -> bf16 (1024 float4, 4/thread)
#pragma unroll
    for (int i = 0; i < 4; ++i) {
      int idx = tid + i * 256;
      int row = idx >> 3, c4 = idx & 7;
      const float4* s = (const float4*)(w2e + (size_t)(h0 + row) * FDIM + kk) + c4;
      cvt_store4(&lB[row * ROWU + c4 * 4], *s);
      if (kk + KT < FDIM) __builtin_prefetch((const void*)(s + 8), 0, 1);
    }
    async_wait0();
    __syncthreads();

    v16bf af[2], bfr[4];
#pragma unroll
    for (int i = 0; i < 2; ++i) af[i] = frag_from_lds(&lA[(Mb + i * 16 + mrow) * ROWU], half);
#pragma unroll
    for (int j = 0; j < 4; ++j) bfr[j] = frag_from_lds(&lB[(Nb + j * 16 + mrow) * ROWU], half);
#pragma unroll
    for (int i = 0; i < 2; ++i)
#pragma unroll
      for (int j = 0; j < 4; ++j)
        acc[i * 4 + j] = __builtin_amdgcn_wmma_f32_16x16x32_bf16(
            false, af[i], false, bfr[j], (short)0, acc[i * 4 + j], false, false);
  }

#pragma unroll
  for (int i = 0; i < 2; ++i)
#pragma unroll
    for (int j = 0; j < 4; ++j) {
      v8f a = acc[i * 4 + j];
      int n = Nb + j * 16 + mrow;
      float* crow = contrib + (size_t)(aBase + Mb + i * 16 + half * 8) * HDIM + (h0 + n);
      if (valid == MT) {
#pragma unroll
        for (int c = 0; c < 8; ++c)
          crow[(size_t)c * HDIM] = sw[Mb + i * 16 + half * 8 + c] * a[c];
      } else {
#pragma unroll
        for (int c = 0; c < 8; ++c) {
          int m = Mb + i * 16 + half * 8 + c;
          if (m < valid) crow[(size_t)c * HDIM] = sw[m] * a[c];
        }
      }
    }
}

// 6) Deterministic combine: each token sums its exactly-two contributions.
__global__ __launch_bounds__(256) void combine_kernel(
    const float* __restrict__ contrib, const int* __restrict__ slot_of,
    float* __restrict__ out)
{
  const int idx = blockIdx.x * 256 + threadIdx.x;  // over T*H/4
  const int t = idx >> 8;                          // H/4 = 256 float4 per row
  const int c = idx & 255;
  const float4* c4 = (const float4*)contrib;
  float4 a = c4[(size_t)slot_of[t * 2 + 0] * 256 + c];
  float4 b = c4[(size_t)slot_of[t * 2 + 1] * 256 + c];
  float4 o; o.x = a.x + b.x; o.y = a.y + b.y; o.z = a.z + b.z; o.w = a.w + b.w;
  ((float4*)out)[idx] = o;
}

// ---------------------------------------------------------------------------
extern "C" void kernel_launch(void* const* d_in, const int* in_sizes, int n_in,
                              void* d_out, int out_size, void* d_ws, size_t ws_size,
                              hipStream_t stream)
{
  // setup_inputs() dict order: hidden_states, gate_w, w1, w3, w2
  const float* hs = (const float*)d_in[0];
  const float* gw = (const float*)d_in[1];
  const float* w1 = (const float*)d_in[2];
  const float* w3 = (const float*)d_in[3];
  const float* w2 = (const float*)d_in[4];
  float* out = (float*)d_out;

  char* ws = (char*)d_ws;
  size_t o = 0;
  auto alloc = [&](size_t bytes) -> char* {
    char* p = ws + o;
    o = (o + bytes + 255) & ~(size_t)255;
    return p;
  };
  int*   cnt      = (int*)  alloc(NEXP * 4);
  int*   cur      = (int*)  alloc(NEXP * 4);
  int*   off      = (int*)  alloc((NEXP + 1) * 4);
  int*   ti       = (int*)  alloc((size_t)T_TOK * TOPK * 4);
  float* tw       = (float*)alloc((size_t)T_TOK * TOPK * 4);
  int*   slot_of  = (int*)  alloc((size_t)T_TOK * TOPK * 4);
  float* wslot    = (float*)alloc((size_t)(T_TOK * TOPK + MT) * 4);
  int*   tok_slot = (int*)  alloc((size_t)(T_TOK * TOPK + MT) * 4);
  bf16*  act      = (bf16*) alloc((size_t)(T_TOK * TOPK + MT) * FDIM * 2);
  float* contrib  = (float*)alloc((size_t)(T_TOK * TOPK + MT) * HDIM * 4);

  (void)hipMemsetAsync(cnt, 0, NEXP * 4, stream);
  (void)hipMemsetAsync(cur, 0, NEXP * 4, stream);

  gate_kernel   <<<T_TOK / 8, 256, 0, stream>>>(hs, gw, cnt, ti, tw);
  offsets_kernel<<<1, 32, 0, stream>>>(cnt, off);
  build_kernel  <<<T_TOK / 256, 256, 0, stream>>>(ti, tw, off, cur, slot_of, wslot, tok_slot);

  dim3 g1(FDIM / NT, T_TOK / MT, NEXP);   // 28 x 32 x 8, early-exit past per-expert count
  ffn1_kernel<<<g1, 256, 0, stream>>>(hs, w1, w3, cnt, off, tok_slot, act);

  dim3 g2(HDIM / NT, T_TOK / MT, NEXP);   // 8 x 32 x 8
  ffn2_kernel<<<g2, 256, 0, stream>>>(act, w2, cnt, off, wslot, contrib);

  combine_kernel<<<(T_TOK * HDIM / 4) / 256, 256, 0, stream>>>(contrib, slot_of, out);
}
